// SeqCondAttention_60851096649908
// MI455X (gfx1250) — matrix operand order repaired and checked by reading, not compiled, not run on hardware
//
#include <hip/hip_runtime.h>
#include <hip/hip_bf16.h>

// ---------------------------------------------------------------------------
// SeqCondAttention forward for MI455X (gfx1250, wave32, WMMA).
//  - bf16 WMMA (v_wmma_f32_16x16x32_bf16) for all four large GEMMs, fp32 accum
//  - big GEMMs: LDS-staged 128x128 block tiles, double-buffered, staged with
//    GLOBAL_LOAD_ASYNC_TO_LDS_B128 (ASYNCcnt) when the toolchain exposes it
//  - chunked 3-pass parallel scan for the linear recurrence
// ---------------------------------------------------------------------------

typedef __bf16 bf16;
typedef __attribute__((ext_vector_type(8)))  bf16  v8bf;
typedef __attribute__((ext_vector_type(16))) bf16  v16bf;
typedef __attribute__((ext_vector_type(8)))  float v8f;
typedef int v4i __attribute__((vector_size(16)));   // int4, matches builtin proto

#define TB        4
#define TL        4096
#define TTOK      16384          // B*L
#define TD        768
#define TK        12
#define TH        64
#define DIM_CONV  1548
#define DIM_MEMT  780
#define DIM_TOT   1752
#define NZ        1792           // z_all row stride (padded N for GEMM1)
#define CHN       1536           // 2*FLAT scan channels
#define NCHUNK    64
#define CLEN      64
#define YD        2304           // K*3*H
#define SWH       384
#define SW        4608           // K*SWH

// block-GEMM tiling
#define BM        128
#define BN        128
#define BK        32
#define LDSW      40             // padded K-stride in LDS (breaks bank conflicts)

#if defined(__HIP_DEVICE_COMPILE__) && \
    __has_builtin(__builtin_amdgcn_global_load_async_to_lds_b128)
#define HAVE_ASYNC_LDS 1
#endif

__device__ __forceinline__ float sigm(float x)  { return 1.0f / (1.0f + expf(-x)); }
__device__ __forceinline__ float siluf(float x) { return x / (1.0f + expf(-x)); }
__device__ __forceinline__ float clampf(float x, float lo, float hi) {
  return fminf(fmaxf(x, lo), hi);
}

__device__ __forceinline__ v16bf cat16(v8bf lo, v8bf hi) {
  return __builtin_shufflevector(lo, hi, 0,1,2,3,4,5,6,7,8,9,10,11,12,13,14,15);
}

// 16B global -> LDS copy; async (ASYNCcnt) when available.
__device__ __forceinline__ void cp_b128(bf16* lds_dst, const bf16* gsrc) {
#ifdef HAVE_ASYNC_LDS
  __builtin_amdgcn_global_load_async_to_lds_b128(
      (__attribute__((address_space(1))) v4i*)(v4i*)(void*)const_cast<bf16*>(gsrc),
      (__attribute__((address_space(3))) v4i*)(v4i*)(void*)lds_dst,
      0, 0);
#else
  *(v8bf*)lds_dst = *(const v8bf*)gsrc;
#endif
}

__device__ __forceinline__ void cp_wait() {
#ifdef HAVE_ASYNC_LDS
#if __has_builtin(__builtin_amdgcn_s_wait_asynccnt)
  __builtin_amdgcn_s_wait_asynccnt(0);
#else
  asm volatile("s_wait_asynccnt 0x0" ::: "memory");
#endif
#endif
}

// -------------------- WMMA fragment loaders (direct / LDS) -----------------
// A (16x32 bf16): lane = m + 16*half;
//                 elems 0..15 -> K = half*16 + e (contiguous 32B).
__device__ __forceinline__ v16bf ldA(const bf16* __restrict__ A, int lda,
                                     int k0, int lane) {
  const int row  = lane & 15;
  const int half = lane >> 4;
  const bf16* p  = A + (long)row * lda + k0 + half * 16;
  return cat16(*(const v8bf*)(p), *(const v8bf*)(p + 8));
}
// B (32x16 bf16, col-major source Bt[n*ldb + k]): lane = n + 16*half;
//                 elems 0..15 -> K = half*16 + e (contiguous 32B).
__device__ __forceinline__ v16bf ldB(const bf16* __restrict__ Bt, int ldb,
                                     int k0, int lane) {
  const int col  = lane & 15;
  const int half = lane >> 4;
  const bf16* p  = Bt + (long)col * ldb + k0 + half * 16;
  return cat16(*(const v8bf*)(p), *(const v8bf*)(p + 8));
}

// -------------------- LDS-staged block GEMM: C = A * B ---------------------
// 256 threads = 8 waves -> 128(M) x 128(N) tile; wave (wr,wc) owns 32x64.
// Requires: M % 128 == 0, N % 128 == 0, Kd % 32 == 0.
__global__ __launch_bounds__(256)
void wmma_gemm_lds(const bf16* __restrict__ A, int lda,
                   const bf16* __restrict__ Bt, int ldb,
                   float* __restrict__ C, int ldc,
                   int N, int Kd) {
  __shared__ bf16 As[2][BM * LDSW];
  __shared__ bf16 Bs[2][BN * LDSW];

  const int tid  = threadIdx.x;
  const int lane = tid & 31;
  const int wv   = tid >> 5;
  const int wr   = wv >> 1;                 // 0..3 : 32-row band
  const int wc   = wv & 1;                  // 0..1 : 64-col band
  const int ntn  = N >> 7;
  const int bm   = blockIdx.x / ntn;
  const int bn   = blockIdx.x % ntn;

  const bf16* Ag = A  + (long)bm * BM * lda;
  const bf16* Bg = Bt + (long)bn * BN * ldb;

  // stage one 128x32 A tile + 128x32 B tile: 512 x 16B each, 4 copies/thread
  auto stage = [&](int p, int k0) {
#pragma unroll
    for (int s = 0; s < 2; ++s) {
      const int j  = tid + s * 256;         // 0..511
      const int r  = j >> 2;
      const int kk = (j & 3) * 8;
      cp_b128(&As[p][r * LDSW + kk], Ag + (long)r * lda + k0 + kk);
      cp_b128(&Bs[p][r * LDSW + kk], Bg + (long)r * ldb + k0 + kk);
    }
  };

  v8f acc[2][4] = {};
  const int nk = Kd / BK;
  stage(0, 0);
  for (int ks = 0; ks < nk; ++ks) {
    cp_wait();                               // own wave's async fills done
    __syncthreads();                         // tile (ks&1) visible block-wide
    if (ks + 1 < nk) stage((ks + 1) & 1, (ks + 1) * BK);

    const bf16* as = As[ks & 1];
    const bf16* bs = Bs[ks & 1];
    const int half = lane >> 4;
    const int mrow = wr * 32 + (lane & 15);

    const bf16* pa0 = as + mrow * LDSW + half * 16;
    v16bf a0 = cat16(*(const v8bf*)(pa0), *(const v8bf*)(pa0 + 8));
    const bf16* pa1 = pa0 + 16 * LDSW;
    v16bf a1 = cat16(*(const v8bf*)(pa1), *(const v8bf*)(pa1 + 8));

#pragma unroll
    for (int ni = 0; ni < 4; ++ni) {
      const int cb = wc * 64 + ni * 16 + (lane & 15);
      const bf16* pb = bs + cb * LDSW + half * 16;
      v16bf b = cat16(*(const v8bf*)(pb), *(const v8bf*)(pb + 8));
      acc[0][ni] = __builtin_amdgcn_wmma_f32_16x16x32_bf16(false, a0, false, b, (short)0, acc[0][ni], false, false);
      acc[1][ni] = __builtin_amdgcn_wmma_f32_16x16x32_bf16(false, a1, false, b, (short)0, acc[1][ni], false, false);
    }
  }

  // C/D layout: VGPR r, lanes 0-15 -> M=r, lanes 16-31 -> M=8+r; N = lane&15.
#pragma unroll
  for (int mi = 0; mi < 2; ++mi) {
    const int row = bm * BM + wr * 32 + mi * 16 + (lane >> 4) * 8;
#pragma unroll
    for (int ni = 0; ni < 4; ++ni) {
      const int col = bn * BN + wc * 64 + ni * 16 + (lane & 15);
#pragma unroll
      for (int r = 0; r < 8; ++r)
        C[(long)(row + r) * ldc + col] = acc[mi][ni][r];
    }
  }
}

// -------------------- fused readout GEMM (spec + skip) ---------------------
// tile id -> (mt in [0,1024), k in [0,12), nc in [0,6)). K = 128 then 192.
__global__ __launch_bounds__(256)
void readout_gemm(const bf16* __restrict__ ocn,   // [T][1536] bf16
                  const bf16* __restrict__ cskip, // [T][192]  bf16
                  const bf16* __restrict__ Wr,    // [12][384][128] col-major/k
                  const bf16* __restrict__ Wsk,   // [4608][192]    col-major
                  float* __restrict__ ytmp)       // [T][4608]
{
  const int lane = threadIdx.x & 31;
  const long tile = (long)blockIdx.x * 8 + (threadIdx.x >> 5);
  if (tile >= (long)1024 * TK * 6) return;   // uniform per wave
  const int nc = (int)(tile % 6);
  const long r2 = tile / 6;
  const int k  = (int)(r2 % TK);
  const int mt = (int)(r2 / TK);

  const bf16* A1 = ocn   + (long)mt * 16 * CHN + k * 128;
  const bf16* B1 = Wr    + ((long)k * SWH + nc * 64) * 128;
  const bf16* A2 = cskip + (long)mt * 16 * 192;
  const bf16* B2 = Wsk   + ((long)k * SWH + nc * 64) * 192;

  v8f c0 = {}, c1 = {}, c2 = {}, c3 = {};
#pragma unroll
  for (int k0 = 0; k0 < 128; k0 += 32) {   // spectral path, K = 2H = 128
    v16bf a  = ldA(A1, CHN, k0, lane);
    v16bf b0 = ldB(B1,             128, k0, lane);
    v16bf b1 = ldB(B1 + 16l * 128, 128, k0, lane);
    v16bf b2 = ldB(B1 + 32l * 128, 128, k0, lane);
    v16bf b3 = ldB(B1 + 48l * 128, 128, k0, lane);
    c0 = __builtin_amdgcn_wmma_f32_16x16x32_bf16(false, a, false, b0, (short)0, c0, false, false);
    c1 = __builtin_amdgcn_wmma_f32_16x16x32_bf16(false, a, false, b1, (short)0, c1, false, false);
    c2 = __builtin_amdgcn_wmma_f32_16x16x32_bf16(false, a, false, b2, (short)0, c2, false, false);
    c3 = __builtin_amdgcn_wmma_f32_16x16x32_bf16(false, a, false, b3, (short)0, c3, false, false);
  }
#pragma unroll
  for (int k0 = 0; k0 < 192; k0 += 32) {   // skip path, K = DIM_SKIP = 192
    v16bf a  = ldA(A2, 192, k0, lane);
    v16bf b0 = ldB(B2,             192, k0, lane);
    v16bf b1 = ldB(B2 + 16l * 192, 192, k0, lane);
    v16bf b2 = ldB(B2 + 32l * 192, 192, k0, lane);
    v16bf b3 = ldB(B2 + 48l * 192, 192, k0, lane);
    c0 = __builtin_amdgcn_wmma_f32_16x16x32_bf16(false, a, false, b0, (short)0, c0, false, false);
    c1 = __builtin_amdgcn_wmma_f32_16x16x32_bf16(false, a, false, b1, (short)0, c1, false, false);
    c2 = __builtin_amdgcn_wmma_f32_16x16x32_bf16(false, a, false, b2, (short)0, c2, false, false);
    c3 = __builtin_amdgcn_wmma_f32_16x16x32_bf16(false, a, false, b3, (short)0, c3, false, false);
  }
  const int row = mt * 16 + (lane >> 4) * 8;
  const int col = k * SWH + nc * 64 + (lane & 15);
#pragma unroll
  for (int r = 0; r < 8; ++r) {
    long o = (long)(row + r) * SW + col;
    ytmp[o]      = c0[r];
    ytmp[o + 16] = c1[r];
    ytmp[o + 32] = c2[r];
    ytmp[o + 48] = c3[r];
  }
}

// -------------------- conversions ------------------------------------------
__global__ __launch_bounds__(256)
void cvt_rowmajor(const float* __restrict__ src, bf16* __restrict__ dst, long n) {
  for (long i = (long)blockIdx.x * blockDim.x + threadIdx.x; i < n;
       i += (long)gridDim.x * blockDim.x)
    dst[i] = (bf16)src[i];
}

// src row-major [Kd][N]  ->  dst col-major [Npad][Kd] (zero pad n >= N)
__global__ __launch_bounds__(256)
void cvt_colmajor(const float* __restrict__ src, bf16* __restrict__ dst,
                  int Kd, int N, int Npad) {
  const long total = (long)Npad * Kd;
  for (long i = (long)blockIdx.x * blockDim.x + threadIdx.x; i < total;
       i += (long)gridDim.x * blockDim.x) {
    int f = (int)(i % Kd);
    long n = i / Kd;
    dst[i] = (n < N) ? (bf16)src[(long)f * N + n] : (bf16)0.0f;
  }
}

// -------------------- per-token prep ---------------------------------------
__global__ __launch_bounds__(256)
void token_prep(const float* __restrict__ z_all,    // [T][NZ]
                const float* __restrict__ conv_w,   // [4][1548]
                const float* __restrict__ wk,       // [64]
                const float* __restrict__ wq,       // [768]
                const float* __restrict__ theta_raw,// [768] (K,H)
                const float* __restrict__ dt_scale, // [12]
                const float* __restrict__ dt_bias,  // [12]
                const float* __restrict__ log_A,    // [12]
                const float* __restrict__ phase,    // [12]
                float* __restrict__ stack,          // [T][1536]
                float* __restrict__ dec12,          // [T][12]
                float* __restrict__ qbuf,           // [T][768]
                bf16*  __restrict__ cskip)          // [T][192]
{
  const int t   = blockIdx.x;
  const int b   = t >> 12;
  const int l   = t & (TL - 1);
  const int tid = threadIdx.x;

  __shared__ float sc[DIM_CONV];
  __shared__ float red[13];   // [0..11] per-k sumsq over H, [12] q sumsq
  __shared__ float dtk[12];

  if (tid < 13) red[tid] = 0.0f;
  __syncthreads();

  for (int c = tid; c < DIM_CONV; c += 256) {
    float v = 0.0f;
#pragma unroll
    for (int j = 0; j < 4; ++j) {
      int lsrc = l + j - 3;
      if (lsrc >= 0)
        v += conv_w[j * DIM_CONV + c] * z_all[((long)(b * TL + lsrc)) * NZ + c];
    }
    v = siluf(v);
    sc[c] = v;
    if (c < 768)            atomicAdd(&red[c >> 6], v * v);
    else if (c >= DIM_MEMT) atomicAdd(&red[12], v * v);
  }
  for (int c = tid; c < 192; c += 256)
    cskip[(long)t * 192 + c] = (bf16)z_all[(long)t * NZ + DIM_CONV + c];
  __syncthreads();

  {
    float qs = rsqrtf(red[12] * (1.0f / 768.0f) + 1e-6f);
    for (int c = tid; c < 768; c += 256)
      qbuf[(long)t * 768 + c] = sc[DIM_MEMT + c] * qs * wq[c];
  }
  if (tid < 12) {
    int k = tid;
    float u  = clampf(dt_scale[k] * sc[768 + k] + dt_bias[k], -20.0f, 20.0f);
    float dt = clampf(log1pf(expf(u)), 0.001f, 0.1f);
    dtk[k]   = dt;
    float ad = clampf(-expf(log_A[k]) * dt, -20.0f, 0.0f);
    dec12[(long)t * 12 + k] = expf(ad);
  }
  __syncthreads();

  for (int c = tid; c < 768; c += 256) {
    int k = c >> 6, h = c & 63;
    float kf  = sc[c] * rsqrtf(red[k] * (1.0f / 64.0f) + 1e-6f) * wk[h];
    float ks  = kf * phase[k];
    float th  = 0.001f + 2.999f * sigm(theta_raw[c]);
    float phi = ks / (1.0f + fabsf(ks)) * th;
    float kvw = kf * dtk[k];
    stack[(long)t * CHN + c]       = kvw * cosf(phi);
    stack[(long)t * CHN + 768 + c] = kvw * sinf(phi);
  }
}

// -------------------- chunked scan -----------------------------------------
__global__ __launch_bounds__(256)
void scan_pass1(const float* __restrict__ stack, const float* __restrict__ dec12,
                float* __restrict__ cA, float* __restrict__ cB) {
  long id = (long)blockIdx.x * 256 + threadIdx.x;   // B*NCHUNK*1536
  if (id >= (long)TB * NCHUNK * CHN) return;
  const int  c = (int)(id % CHN);
  const long r = id / CHN;
  const int  j = (int)(r % NCHUNK);
  const int  b = (int)(r / NCHUNK);
  const int  k = (c & 767) >> 6;
  float Aacc = 1.0f, s = 0.0f;
  const long tbase = (long)b * TL + j * CLEN;
  for (int i = 0; i < CLEN; ++i) {
    long t = tbase + i;
    float d = dec12[t * 12 + k];
    s = d * s + stack[t * CHN + c];
    Aacc *= d;
  }
  cA[id] = Aacc;
  cB[id] = s;
}

__global__ __launch_bounds__(256)
void scan_pass2(const float* __restrict__ cA, const float* __restrict__ cB,
                float* __restrict__ cin) {
  int id = blockIdx.x * 256 + threadIdx.x;          // B*1536
  if (id >= TB * CHN) return;
  const int c = id % CHN, b = id / CHN;
  float s = 0.0f;
  for (int j = 0; j < NCHUNK; ++j) {
    long o = ((long)(b * NCHUNK + j)) * CHN + c;
    cin[o] = s;
    s = cA[o] * s + cB[o];
  }
}

__global__ __launch_bounds__(256)
void scan_pass3(const float* __restrict__ stack, const float* __restrict__ dec12,
                const float* __restrict__ cin, const float* __restrict__ qbuf,
                const float* __restrict__ wint_raw, float* __restrict__ oc) {
  long id = (long)blockIdx.x * 256 + threadIdx.x;   // B*NCHUNK*768
  if (id >= (long)TB * NCHUNK * 768) return;
  const int  c  = (int)(id % 768);
  const long r  = id / 768;
  const int  j  = (int)(r % NCHUNK);
  const int  b  = (int)(r / NCHUNK);
  const int  k  = c >> 6, h = c & 63, kq = k >> 1;
  const float w = expf(clampf(wint_raw[c], -5.0f, 5.0f));

  const long co = ((long)(b * NCHUNK + j)) * CHN;
  float sre = cin[co + c];
  float sim = cin[co + 768 + c];
  const long tbase = (long)b * TL + j * CLEN;
  for (int i = 0; i < CLEN; ++i) {
    long t = tbase + i;
    float d = dec12[t * 12 + k];
    sre = d * sre + stack[t * CHN + c];
    sim = d * sim + stack[t * CHN + 768 + c];
    float qr = qbuf[t * 768 + kq * 128 + h * 2];
    float qi = qbuf[t * 768 + kq * 128 + h * 2 + 1];
    oc[t * CHN + k * 128 + h]      = (sre * qr + sim * qi) * w;
    oc[t * CHN + k * 128 + 64 + h] = (sim * qr - sre * qi) * w;
  }
}

// -------------------- gate + group-RMS -> bf16 -----------------------------
__global__ __launch_bounds__(256)
void gate_norm(const float* __restrict__ z_all, const float* __restrict__ oc,
               const float* __restrict__ W_gate, const float* __restrict__ gn_w,
               bf16* __restrict__ ocn) {
  const int t = blockIdx.x, tid = threadIdx.x;
  __shared__ float gl[12];
  __shared__ float tmp[CHN];
  __shared__ float ssq;
  if (tid == 0) ssq = 0.0f;
  if (tid < 12) gl[tid] = z_all[(long)t * NZ + (DIM_TOT - TK) + tid];
  __syncthreads();

  float acc = 0.0f;
  for (int c = tid; c < CHN; c += 256) {
    float dot = 0.0f;
#pragma unroll
    for (int k = 0; k < 12; ++k) dot += gl[k] * W_gate[k * CHN + c];
    float v = oc[(long)t * CHN + c] * sigm(dot);
    tmp[c] = v;
    acc += v * v;
  }
  atomicAdd(&ssq, acc);
  __syncthreads();
  const float scl = rsqrtf(ssq * (1.0f / (float)CHN) + 1e-6f);
  for (int c = tid; c < CHN; c += 256)
    ocn[(long)t * CHN + c] = (bf16)(tmp[c] * scl * gn_w[c]);
}

// -------------------- val * silu(gate) -> bf16 y ---------------------------
__global__ __launch_bounds__(256)
void ygate(const float* __restrict__ ytmp, bf16* __restrict__ ybf) {
  const long total = (long)TTOK * YD;
  for (long i = (long)blockIdx.x * 256 + threadIdx.x; i < total;
       i += (long)gridDim.x * 256) {
    long t = i / YD;
    int  r = (int)(i % YD);
    int  k = r / 192, j = r % 192;
    float a = ytmp[t * SW + k * SWH + j];
    float g = ytmp[t * SW + k * SWH + 192 + j];
    ybf[i] = (bf16)(a * siluf(g));
  }
}

// ---------------------------------------------------------------------------
extern "C" void kernel_launch(void* const* d_in, const int* in_sizes, int n_in,
                              void* d_out, int out_size, void* d_ws, size_t ws_size,
                              hipStream_t stream) {
  const float* x         = (const float*)d_in[0];
  const float* W_in      = (const float*)d_in[1];
  const float* conv_w    = (const float*)d_in[2];
  const float* wk        = (const float*)d_in[3];
  const float* wq        = (const float*)d_in[4];
  const float* theta_raw = (const float*)d_in[5];
  const float* w_int_raw = (const float*)d_in[6];
  const float* dt_scale  = (const float*)d_in[7];
  const float* dt_bias   = (const float*)d_in[8];
  const float* log_A     = (const float*)d_in[9];
  const float* phase     = (const float*)d_in[10];
  const float* W_gate    = (const float*)d_in[11];
  const float* gn_w      = (const float*)d_in[12];
  const float* W_ro      = (const float*)d_in[13];
  const float* W_skip    = (const float*)d_in[14];
  const float* W_out     = (const float*)d_in[15];
  float* out = (float*)d_out;

  char* bump = (char*)d_ws;
  auto alloc = [&](size_t bytes) -> void* {
    void* r = (void*)bump;
    bump += (bytes + 255) & ~(size_t)255;
    return r;
  };
  bf16*  xbf   = (bf16*) alloc((size_t)TTOK * TD * 2);
  bf16*  Wi    = (bf16*) alloc((size_t)NZ * TD * 2);          // [1792][768] cm
  bf16*  Wr    = (bf16*) alloc((size_t)TK * SWH * 128 * 2);   // per-k [384][128] cm
  bf16*  Wsk   = (bf16*) alloc((size_t)SW * 192 * 2);         // [4608][192] cm
  bf16*  Wo    = (bf16*) alloc((size_t)TD * YD * 2);          // [768][2304] cm
  float* z_all = (float*)alloc((size_t)TTOK * NZ * 4);
  bf16*  cskip = (bf16*) alloc((size_t)TTOK * 192 * 2);
  float* qbuf  = (float*)alloc((size_t)TTOK * TD * 4);
  float* stack = (float*)alloc((size_t)TTOK * CHN * 4);
  float* dec12 = (float*)alloc((size_t)TTOK * TK * 4);
  float* cA    = (float*)alloc((size_t)TB * NCHUNK * CHN * 4);
  float* cB    = (float*)alloc((size_t)TB * NCHUNK * CHN * 4);
  float* cin   = (float*)alloc((size_t)TB * NCHUNK * CHN * 4);
  float* oc    = (float*)alloc((size_t)TTOK * CHN * 4);
  bf16*  ocn   = (bf16*) alloc((size_t)TTOK * CHN * 2);
  float* ytmp  = (float*)alloc((size_t)TTOK * SW * 4);
  bf16*  ybf   = (bf16*) alloc((size_t)TTOK * YD * 2);

  // ---- weight / activation conversions to bf16 (col-major weights) ----
  cvt_rowmajor<<<4096, 256, 0, stream>>>(x, xbf, (long)TTOK * TD);
  cvt_colmajor<<<2688, 256, 0, stream>>>(W_in, Wi, TD, DIM_TOT, NZ);
  for (int k = 0; k < TK; ++k)
    cvt_colmajor<<<192, 256, 0, stream>>>(W_ro + (size_t)k * 128 * SWH,
                                          Wr + (size_t)k * SWH * 128, 128, SWH, SWH);
  cvt_colmajor<<<3456, 256, 0, stream>>>(W_skip, Wsk, 192, SW, SW);
  cvt_colmajor<<<6912, 256, 0, stream>>>(W_out, Wo, YD, TD, TD);

  // ---- 1) z_all = x @ W_in  (16384 x 768 x 1792-padded), LDS-staged ----
  wmma_gemm_lds<<<(TTOK / BM) * (NZ / BN), 256, 0, stream>>>(
      xbf, TD, Wi, TD, z_all, NZ, NZ, TD);

  // ---- 2) conv + SiLU + norms + dt/phase math ----
  token_prep<<<TTOK, 256, 0, stream>>>(z_all, conv_w, wk, wq, theta_raw,
                                       dt_scale, dt_bias, log_A, phase,
                                       stack, dec12, qbuf, cskip);

  // ---- 3) chunked linear-recurrence scan + fused q-match ----
  scan_pass1<<<(TB * NCHUNK * CHN) / 256, 256, 0, stream>>>(stack, dec12, cA, cB);
  scan_pass2<<<(TB * CHN + 255) / 256, 256, 0, stream>>>(cA, cB, cin);
  scan_pass3<<<(TB * NCHUNK * 768) / 256, 256, 0, stream>>>(stack, dec12, cin,
                                                            qbuf, w_int_raw, oc);

  // ---- 4) gating + group RMS -> bf16 ----
  gate_norm<<<TTOK, 256, 0, stream>>>(z_all, oc, W_gate, gn_w, ocn);

  // ---- 5) fused readout: spec einsum + skip GEMM ----
  readout_gemm<<<9216, 256, 0, stream>>>(ocn, cskip, Wr, Wsk, ytmp);

  // ---- 6) val * silu(gate) ----
  ygate<<<8192, 256, 0, stream>>>(ytmp, ybf);

  // ---- 7) out = y @ W_out  (16384 x 2304 x 768), LDS-staged ----
  wmma_gemm_lds<<<(TTOK / BM) * (TD / BN), 256, 0, stream>>>(
      ybf, YD, Wo, YD, out, TD, TD, YD);
}
